// TracksMoE_27745488732223
// MI455X (gfx1250) — matrix operand order repaired and checked by reading, not compiled, not run on hardware
//
#include <hip/hip_runtime.h>
#include <stdint.h>

// Problem constants
#define B_    2
#define S_    896
#define DIM_  1536
#define E_    8
#define TR_   1643
#define N_    (B_*TR_)     // 3286 tokens
#define NPAD_ 3296         // 206 * 16
#define H_    1792         // 2*S
#define NT_   206          // 16-row tiles over NPAD_
#define XS_   453          // LDS x-tile row stride in u32 (448 data + pad, odd: no bank conflict)
#define HS_   905          // LDS h-tile row stride in u32 (896 data + pad, odd)

typedef __attribute__((ext_vector_type(16))) __bf16 v16bf;
typedef __attribute__((ext_vector_type(8)))  float  v8f;

union FragBF { v16bf v; uint32_t u[8]; uint4 q[2]; };

__device__ __forceinline__ uint16_t f2bf(float f) {
  uint32_t u = __float_as_uint(f);
  uint32_t r = 0x7FFFu + ((u >> 16) & 1u);   // round-to-nearest-even
  return (uint16_t)((u + r) >> 16);
}
__device__ __forceinline__ float bf2f(uint16_t h) {
  return __uint_as_float(((uint32_t)h) << 16);
}

__device__ __forceinline__ void loadB(FragBF& b, const uint32_t* p) {
  b.q[0] = *(const uint4*)(p);
  b.q[1] = *(const uint4*)(p + 4);
}
// A-fragment (16-bit 16x32): lane row = lane&15; K = (lane>>4)*16 + 2v
__device__ __forceinline__ void loadA(FragBF& a, const uint32_t* row, int kb) {
  #pragma unroll
  for (int v = 0; v < 8; ++v) {
    int k = kb + (kb & 8) + 2 * v;
    a.u[v] = row[k >> 1];
  }
}
#define WMMA_BF16(A, Bf, C) \
  __builtin_amdgcn_wmma_f32_16x16x32_bf16(false, (A).v, false, (Bf).v, (short)0, (C), false, false)

// ---------------- zero workspace accumulators ----------------
__global__ void zero_ws(float* __restrict__ yacc, float* __restrict__ gf,
                        float* __restrict__ stats) {
  int idx = blockIdx.x * 256 + threadIdx.x;
  if (idx < NPAD_ * S_) yacc[idx] = 0.f;
  if (idx < NPAD_ * 8)  gf[idx]   = 0.f;
  if (idx < 64)         stats[idx] = 0.f;
}

// ---------------- pack W1 (E,S,H) into WMMA B-fragment order ----------------
// fragment u32 index: (((e*112 + hc)*28 + kc)*32 + lane)*8 + v
// B layout (16-bit, 32x16): lane group l16 holds K = l16*16 + 2v, 2v+1; col = hc*16 + (lane&15)
__global__ void pack_w1(const float* __restrict__ W1, uint32_t* __restrict__ W1p) {
  int idx = blockIdx.x * 256 + threadIdx.x;
  int v    = idx & 7;
  int lane = (idx >> 3) & 31;
  int r2   = idx >> 8;
  int kc = r2 % 28; r2 /= 28;
  int hc = r2 % 112;
  int e  = r2 / 112;
  int k0  = kc * 32 + (lane >> 4) * 16 + 2 * v;     // k over S
  int col = hc * 16 + (lane & 15);                  // col over H
  size_t lo = ((size_t)e * S_ + k0) * H_ + col;
  W1p[idx] = (uint32_t)f2bf(W1[lo]) | ((uint32_t)f2bf(W1[lo + H_]) << 16);
}

// ---------------- pack W2 (E,H,S) into WMMA B-fragment order ----------------
// fragment u32 index: (((e*56 + sc)*56 + kc)*32 + lane)*8 + v ; k over H, col over S
__global__ void pack_w2(const float* __restrict__ W2, uint32_t* __restrict__ W2p) {
  int idx = blockIdx.x * 256 + threadIdx.x;
  int v    = idx & 7;
  int lane = (idx >> 3) & 31;
  int r2   = idx >> 8;
  int kc = r2 % 56; r2 /= 56;
  int sc = r2 % 56;
  int e  = r2 / 56;
  int k0  = kc * 32 + (lane >> 4) * 16 + 2 * v;
  int col = sc * 16 + (lane & 15);
  size_t lo = ((size_t)e * H_ + k0) * S_ + col;
  W2p[idx] = (uint32_t)f2bf(W2[lo]) | ((uint32_t)f2bf(W2[lo + S_]) << 16);
}

// ---------------- gather o = swapaxes(out,1,2)[:, index, :] -> bf16 rows ----------------
__global__ void gather_obf(const float* __restrict__ out, const int* __restrict__ index,
                           uint32_t* __restrict__ obf32) {
  int idx = blockIdx.x * 256 + threadIdx.x;   // NPAD_*448
  int n = idx / 448, c = idx - n * 448;
  uint32_t val = 0u;
  if (n < N_) {
    int b = n / TR_, t = n % TR_;
    int col = index[t];
    const float* p = out + ((size_t)b * S_ + 2 * c) * TR_ + col;
    val = (uint32_t)f2bf(p[0]) | ((uint32_t)f2bf(p[TR_]) << 16);
  }
  obf32[idx] = val;
}

// ---------------- inverse permutation ----------------
__global__ void inv_index(const int* __restrict__ index, int* __restrict__ inv) {
  int t = blockIdx.x * 256 + threadIdx.x;
  if (t < TR_) inv[index[t]] = t;
}

// ---------------- xmean = x.mean(axis=1), (B, 2*DIM) ----------------
__global__ void xmean_k(const float* __restrict__ x, float* __restrict__ xmean) {
  int idx = blockIdx.x * 256 + threadIdx.x;   // 6144
  int b = idx / (2 * DIM_), d = idx % (2 * DIM_);
  const float* p = x + (size_t)b * S_ * 2 * DIM_ + d;
  float s = 0.f;
  for (int t = 0; t < S_; ++t) s += p[(size_t)t * 2 * DIM_];
  xmean[idx] = s * (1.f / S_);
}

// ---------------- small gates: gate_logits, emb_logits, w2, base zloss ----------------
// stats layout: [0]=zloss, [1..32]=all_gates(4x8), [33..48]=w2 (B x 8)
__global__ __launch_bounds__(64)
void small_gates(const float* __restrict__ xmean, const float* __restrict__ Wsel,
                 const float* __restrict__ bsel, const float* __restrict__ emb,
                 const float* __restrict__ Wemb, const float* __restrict__ bemb,
                 float* __restrict__ stats) {
  __shared__ float gl[16];
  __shared__ float el[8];
  int tid = threadIdx.x;
  if (tid < 16) {
    int b = tid >> 3, g = tid & 7;
    float s = bsel[g];
    for (int d = 0; d < 2 * DIM_; ++d) s += xmean[b * 2 * DIM_ + d] * Wsel[d * 8 + g];
    gl[tid] = s;
  } else if (tid < 24) {
    int g = tid - 16;
    float s = bemb[g];
    for (int d = 0; d < DIM_; ++d) s += emb[d] * Wemb[d * 8 + g];
    el[g] = s;
  }
  __syncthreads();
  if (tid == 0) {
    float zl = 0.f;
    for (int i = 0; i < 4; ++i) {
      float e0 = el[2 * i], e1 = el[2 * i + 1];
      zl += (e0 * e0 + e1 * e1) * 0.5f;
      float m = fmaxf(e0, e1);
      float x0 = __expf(e0 - m), x1 = __expf(e1 - m);
      float se = x0 + x1;
      float sm0 = x0 / se, sm1 = x1 / se;
      for (int b = 0; b < 2; ++b) {
        float t0 = gl[b * 8 + 2 * i], t1 = gl[b * 8 + 2 * i + 1];
        zl += (t0 * t0 + t1 * t1) * 0.25f;
        float mm = fmaxf(t0, t1);
        float y0 = __expf(t0 - mm), y1 = __expf(t1 - mm);
        float st = y0 + y1;
        stats[33 + b * 8 + 2 * i]     = 0.5f * (y0 / st + sm0);
        stats[33 + b * 8 + 2 * i + 1] = 0.5f * (y1 / st + sm1);
      }
    }
    atomicAdd(&stats[0], zl);
  }
}

// ---------------- per-token gating: logits, leaky-relu, top-3 softmax -> gf ----------------
__global__ __launch_bounds__(256)
void gating(const uint16_t* __restrict__ obf, const float* __restrict__ temb,
            const float* __restrict__ Wg, const float* __restrict__ bg,
            float* __restrict__ gf, float* __restrict__ stats) {
  __shared__ float red0[256], red1[256];
  int n = blockIdx.x;
  int b = n / TR_, t = n % TR_;
  int i, T;
  if      (t < 228)  { i = 0; T = 228; }
  else if (t < 519)  { i = 1; T = 291; }
  else if (t < 1286) { i = 2; T = 767; }
  else               { i = 3; T = 357; }
  int tid = threadIdx.x;
  int e = tid & 7, part = tid >> 3;                 // 32 parts x 28 cols
  const float* wg0 = Wg + (size_t)(2 * i) * S_ * 8;
  const float* wg1 = wg0 + S_ * 8;
  float a0 = 0.f, a1 = 0.f;
  for (int q = 0; q < 28; ++q) {
    int s = part * 28 + q;
    float xv = bf2f(obf[(size_t)n * S_ + s]) + temb[i * S_ + s];
    a0 += xv * wg0[s * 8 + e];
    a1 += xv * wg1[s * 8 + e];
  }
  red0[tid] = a0; red1[tid] = a1;
  __syncthreads();
  for (int off = 128; off >= 8; off >>= 1) {
    if (tid < off) { red0[tid] += red0[tid + off]; red1[tid] += red1[tid + off]; }
    __syncthreads();
  }
  if (tid == 0) {
    float gacc[8] = {0, 0, 0, 0, 0, 0, 0, 0};
    float zl = 0.f;
    for (int j = 0; j < 2; ++j) {
      int g = 2 * i + j;
      float lg[8], tmp[8];
      for (int k = 0; k < 8; ++k) {
        float v = (j ? red1[k] : red0[k]) + bg[g * 8 + k];
        lg[k] = v >= 0.f ? v : 0.01f * v;     // leaky_relu
        zl += lg[k] * lg[k];
        tmp[k] = lg[k];
      }
      float tv[3]; int ti[3];
      for (int k = 0; k < 3; ++k) {
        int bi = 0; float bv = tmp[0];
        for (int m = 1; m < 8; ++m) if (tmp[m] > bv) { bv = tmp[m]; bi = m; }
        tv[k] = bv; ti[k] = bi; tmp[bi] = -1e30f;
      }
      float e0 = 1.f, e1 = __expf(tv[1] - tv[0]), e2 = __expf(tv[2] - tv[0]);
      float ssum = e0 + e1 + e2;
      float w2j = stats[33 + b * 8 + g];
      gacc[ti[0]] += (e0 / ssum) * w2j;
      gacc[ti[1]] += (e1 / ssum) * w2j;
      gacc[ti[2]] += (e2 / ssum) * w2j;
    }
    atomicAdd(&stats[0], zl / (2.f * (float)T * 8.f));
    for (int k = 0; k < 8; ++k) {
      gf[(size_t)n * 8 + k] = gacc[k];
      atomicAdd(&stats[1 + i * 8 + k], gacc[k]);
    }
  }
}

// ---------------- main MoE: bf16 WMMA, per (16-row tile, expert) ----------------
// Dual output tiles per pass (one A feeds two WMMA chains), B fragments
// ping-pong double-buffered, K-loop kept rolled (#pragma unroll 1) so the
// allocator keeps distinct buffers and loads overlap the WMMA chain.
__global__ __launch_bounds__(256)
void moe_main(const uint32_t* __restrict__ obf32, const uint32_t* __restrict__ W1p,
              const uint32_t* __restrict__ W2p, const float* __restrict__ b1,
              const float* __restrict__ b2, const float* __restrict__ gf,
              float* __restrict__ yacc) {
  extern __shared__ uint32_t smem[];
  uint32_t* xs = smem;                 // 16 x XS_ u32 (x tile, bf16 pairs)
  uint32_t* hs = smem + 16 * XS_;      // 16 x HS_ u32 (h tile, bf16 pairs)
  const int row0 = blockIdx.x * 16;
  const int e    = blockIdx.y;
  const int tid  = threadIdx.x;
  const int lane = tid & 31, wave = tid >> 5;       // 8 waves
  const int l16  = lane >> 4, ln = lane & 15;

  // stage x tile (16 rows x 896 bf16)
  for (int i = tid; i < 16 * 448; i += 256) {
    int r = i / 448, c = i - r * 448;
    xs[r * XS_ + c] = obf32[(size_t)(row0 + r) * 448 + c];
  }
  __syncthreads();

  uint16_t* hs16 = (uint16_t*)hs;
  const uint32_t* xrow = xs + ln * XS_;
  const uint32_t* hrow = hs + ln * HS_;

  // Phase 1: h = relu(x @ W1[e] + b1[e]) -> LDS (16 x 1792 bf16)
  for (int p = 0; p < 7; ++p) {
    int hc0 = wave * 14 + 2 * p, hc1 = hc0 + 1;
    const uint32_t* bp0 = W1p + ((((size_t)e * 112 + hc0) * 28) * 32 + lane) * 8;
    const uint32_t* bp1 = W1p + ((((size_t)e * 112 + hc1) * 28) * 32 + lane) * 8;
    v8f acc0 = {}, acc1 = {};
    FragBF b0a, b0b, b1a, b1b;
    loadB(b0a, bp0); loadB(b1a, bp1);
    #pragma unroll 1
    for (int kc = 0; kc < 26; kc += 2) {
      FragBF a0, a1;
      loadA(a0, xrow, kc * 32 + l16 * 8);
      loadA(a1, xrow, (kc + 1) * 32 + l16 * 8);
      loadB(b0b, bp0 + 256); loadB(b1b, bp1 + 256);
      acc0 = WMMA_BF16(a0, b0a, acc0);
      acc1 = WMMA_BF16(a0, b1a, acc1);
      loadB(b0a, bp0 + 512); loadB(b1a, bp1 + 512);   // prefetch pair kc+2
      acc0 = WMMA_BF16(a1, b0b, acc0);
      acc1 = WMMA_BF16(a1, b1b, acc1);
      bp0 += 512; bp1 += 512;
    }
    { // epilogue: kc = 26, 27 (b*a already loaded)
      FragBF a0, a1;
      loadA(a0, xrow, 26 * 32 + l16 * 8);
      loadA(a1, xrow, 27 * 32 + l16 * 8);
      loadB(b0b, bp0 + 256); loadB(b1b, bp1 + 256);
      acc0 = WMMA_BF16(a0, b0a, acc0);
      acc1 = WMMA_BF16(a0, b1a, acc1);
      acc0 = WMMA_BF16(a1, b0b, acc0);
      acc1 = WMMA_BF16(a1, b1b, acc1);
    }
    int hcol0 = hc0 * 16 + ln, hcol1 = hc1 * 16 + ln;
    float bias0 = b1[e * H_ + hcol0], bias1 = b1[e * H_ + hcol1];
    #pragma unroll
    for (int r = 0; r < 8; ++r) {
      int row = r + l16 * 8;                        // C/D fragment row mapping
      float h0 = acc0[r] + bias0; h0 = h0 > 0.f ? h0 : 0.f;
      float h1 = acc1[r] + bias1; h1 = h1 > 0.f ? h1 : 0.f;
      hs16[row * (2 * HS_) + hcol0] = f2bf(h0);
      hs16[row * (2 * HS_) + hcol1] = f2bf(h1);
    }
  }
  __syncthreads();

  // per-row gate values for this expert (reused across all column tiles)
  float gate[8];
  #pragma unroll
  for (int r = 0; r < 8; ++r)
    gate[r] = gf[(size_t)(row0 + r + l16 * 8) * 8 + e];   // zero on pad rows

  // Phase 2: y += gf[n,e] * (h @ W2[e] + b2[e])
  for (int u = 0; u < 4; ++u) {
    int sc0 = wave * 7 + 2 * u;
    const bool dual = (u < 3);                      // u==3 -> single tile (7th)
    int sc1 = sc0 + 1;
    const uint32_t* bp0 = W2p + ((((size_t)e * 56 + sc0) * 56) * 32 + lane) * 8;
    const uint32_t* bp1 = W2p + ((((size_t)e * 56 + sc1) * 56) * 32 + lane) * 8;
    v8f acc0 = {}, acc1 = {};
    FragBF b0a, b0b, b1a, b1b;
    loadB(b0a, bp0);
    if (dual) loadB(b1a, bp1);
    #pragma unroll 1
    for (int kc = 0; kc < 54; kc += 2) {
      FragBF a0, a1;
      loadA(a0, hrow, kc * 32 + l16 * 8);
      loadA(a1, hrow, (kc + 1) * 32 + l16 * 8);
      loadB(b0b, bp0 + 256);
      if (dual) loadB(b1b, bp1 + 256);
      acc0 = WMMA_BF16(a0, b0a, acc0);
      if (dual) acc1 = WMMA_BF16(a0, b1a, acc1);
      loadB(b0a, bp0 + 512);                        // prefetch pair kc+2
      if (dual) loadB(b1a, bp1 + 512);
      acc0 = WMMA_BF16(a1, b0b, acc0);
      if (dual) acc1 = WMMA_BF16(a1, b1b, acc1);
      bp0 += 512; bp1 += 512;
    }
    { // epilogue: kc = 54, 55
      FragBF a0, a1;
      loadA(a0, hrow, 54 * 32 + l16 * 8);
      loadA(a1, hrow, 55 * 32 + l16 * 8);
      loadB(b0b, bp0 + 256);
      if (dual) loadB(b1b, bp1 + 256);
      acc0 = WMMA_BF16(a0, b0a, acc0);
      if (dual) acc1 = WMMA_BF16(a0, b1a, acc1);
      acc0 = WMMA_BF16(a1, b0b, acc0);
      if (dual) acc1 = WMMA_BF16(a1, b1b, acc1);
    }
    int col0 = sc0 * 16 + ln;
    float bias0 = b2[e * S_ + col0];
    #pragma unroll
    for (int r = 0; r < 8; ++r) {
      int row = row0 + r + l16 * 8;
      atomicAdd(&yacc[(size_t)row * S_ + col0], (acc0[r] + bias0) * gate[r]);
    }
    if (dual) {
      int col1 = sc1 * 16 + ln;
      float bias1 = b2[e * S_ + col1];
      #pragma unroll
      for (int r = 0; r < 8; ++r) {
        int row = row0 + r + l16 * 8;
        atomicAdd(&yacc[(size_t)row * S_ + col1], (acc1[r] + bias1) * gate[r]);
      }
    }
  }
}

// ---------------- scatter + residual into y_out (coalesced stores) ----------------
__global__ void finalize_y(const float* __restrict__ yacc, const float* __restrict__ out,
                           const int* __restrict__ inv, float* __restrict__ dout) {
  int idx = blockIdx.x * 256 + threadIdx.x;   // B_*S_*TR_
  int col = idx % TR_;
  int bs  = idx / TR_;                        // b*S + s
  int b = bs / S_, s = bs - b * S_;
  int t = inv[col];
  dout[idx] = yacc[((size_t)b * TR_ + t) * S_ + s] + out[idx];
}

__global__ void finalize_tail(const float* __restrict__ stats, float* __restrict__ dout) {
  int tid = threadIdx.x;
  const int base = B_ * S_ * TR_;             // 2944256
  if (tid < 32)       dout[base + tid] = stats[1 + tid];          // all_gates (4x8)
  else if (tid == 32) dout[base + 32] = stats[0];                 // zloss
  else if (tid == 33) dout[base + 33] = 0.f;                      // zero scalar
  else if (tid < 50)  dout[base + tid] = stats[33 + (tid - 34)];  // weights (2x8)
}

extern "C" void kernel_launch(void* const* d_in, const int* in_sizes, int n_in,
                              void* d_out, int out_size, void* d_ws, size_t ws_size,
                              hipStream_t stream) {
  (void)in_sizes; (void)n_in; (void)out_size; (void)ws_size;
  const float* x     = (const float*)d_in[0];
  const float* out   = (const float*)d_in[1];
  const float* emb   = (const float*)d_in[2];
  const int*   index = (const int*)d_in[3];
  const float* Wsel  = (const float*)d_in[4];
  const float* bsel  = (const float*)d_in[5];
  const float* Wemb  = (const float*)d_in[6];
  const float* bemb  = (const float*)d_in[7];
  const float* Wg    = (const float*)d_in[8];
  const float* bg    = (const float*)d_in[9];
  const float* temb  = (const float*)d_in[10];
  const float* W1    = (const float*)d_in[11];
  const float* b1    = (const float*)d_in[12];
  const float* W2    = (const float*)d_in[13];
  const float* b2    = (const float*)d_in[14];
  float* dout = (float*)d_out;

  // workspace layout (bytes), total ~69.3 MB
  char* ws = (char*)d_ws;
  float*    yacc  = (float*)(ws);                    // NPAD_*S_*4       = 11,812,864
  float*    gf    = (float*)(ws + 11812864);         // NPAD_*8*4        =    105,472
  float*    xmean = (float*)(ws + 11918336);         // 6144*4           =     24,576
  float*    stats = (float*)(ws + 11942912);         // 64*4 (+pad)      =        256
  int*      inv   = (int*)(ws + 11943168);           // TR_*4 (+pad)     =      6,656
  uint32_t* obf32 = (uint32_t*)(ws + 11949824);      // NPAD_*S_*2       =  5,906,432
  uint32_t* W1p   = (uint32_t*)(ws + 17856256);      // 8*112*28*256*4   = 25,690,112
  uint32_t* W2p   = (uint32_t*)(ws + 43546368);      // 8*56*56*256*4    = 25,690,112

  zero_ws    <<<11536, 256, 0, stream>>>(yacc, gf, stats);
  pack_w1    <<<25088, 256, 0, stream>>>(W1, W1p);
  pack_w2    <<<25088, 256, 0, stream>>>(W2, W2p);
  gather_obf <<<5768,  256, 0, stream>>>(out, index, obf32);
  inv_index  <<<7,     256, 0, stream>>>(index, inv);
  xmean_k    <<<24,    256, 0, stream>>>(x, xmean);
  small_gates<<<1, 64, 0, stream>>>(xmean, Wsel, bsel, emb, Wemb, bemb, stats);
  gating     <<<N_, 256, 0, stream>>>((const uint16_t*)obf32, temb, Wg, bg, gf, stats);

  dim3 grid(NT_, E_);
  size_t shmem = (size_t)(16 * XS_ + 16 * HS_) * 4;  // 86,912 bytes
  moe_main   <<<grid, 256, shmem, stream>>>(obf32, W1p, W2p, b1, b2, gf, yacc);

  finalize_y   <<<11501, 256, 0, stream>>>(yacc, out, inv, dout);
  finalize_tail<<<1, 64, 0, stream>>>(stats, dout);
}
